// Network_27994596835705
// MI455X (gfx1250) — compile-verified, matrix operand order checked by reading, not run on hardware
//
#include <hip/hip_runtime.h>
#include <hip/hip_bf16.h>

// ---------------------------------------------------------------------------
// Graph-transformer forward (B=16, NN=1000, R=250, D=128, H=4, C=128, L=3)
// Dense layers: v_wmma_f32_16x16x32_f16, f16 activation mirrors, A tile staged
// in LDS via global_load_async_to_lds_b128 (ASYNCcnt), fragments via
// ds_load_b128, B via global b128 loads.
// Edge softmax: wave-per-(edge,head), shfl reduction, ordered-uint atomicMax,
// two-pass exp/sum with float atomicAdd.
// ---------------------------------------------------------------------------

#define NB   16
#define NNODE 1000
#define NR   250
#define NX   (NB*NNODE)     // 16000 x-nodes
#define NT   (NB*NR)        // 4000 t-nodes
#define DD   128            // hidden
#define NH   4              // heads
#define CC   128            // per-head dim
#define HC   (NH*CC)        // 512
#define NE   128000         // edges per graph

typedef __attribute__((ext_vector_type(16))) _Float16 v16h;
typedef __attribute__((ext_vector_type(8)))  float    v8f;

union Frag16 { unsigned u[8]; uint4 q[2]; v16h v; };

__device__ __forceinline__ unsigned short f16bits(float f) {
  _Float16 h = (_Float16)f;
  return __builtin_bit_cast(unsigned short, h);
}

// monotonic float <-> uint encoding for atomic segment-max
__device__ __forceinline__ unsigned fenc(float f) {
  int i = __float_as_int(f);
  return (i >= 0) ? ((unsigned)i | 0x80000000u) : ~(unsigned)i;
}
__device__ __forceinline__ float fdec(unsigned u) {
  int i = (u & 0x80000000u) ? (int)(u & 0x7FFFFFFFu) : (int)~u;
  return __int_as_float(i);
}
#define ENC_NEG_INF 0x007FFFFFu   // fenc(-inf)

typedef __attribute__((address_space(3))) unsigned short lds_us_t;

// ---------------------------------------------------------------------------
// WMMA GEMM: Out[M,N] = act(A[M,128] x W + bias).
//   Ah : activations, f16 bits, row-major [M][128]
//   Wt : weights, f16 bits, transposed [N][128]
// Block = 256 thr = 8 waves, tile 16x128; A tile (16x128 f16 = 4KB) staged in
// LDS once via async-to-LDS DMA, each wave owns one 16x16 WMMA tile;
// K-loop = 4 x 32.  ACT: 0=none 1=relu.  ADDTO: Out += result.
// HOUT: also write f16 mirror to Outh.
// ---------------------------------------------------------------------------
template <int ACT, int ADDTO, int HOUT>
__global__ __launch_bounds__(256)
void k_gemm(const unsigned short* __restrict__ Ah,
            const unsigned short* __restrict__ Wt,
            const float* __restrict__ bias,
            float* __restrict__ Out, unsigned short* __restrict__ Outh,
            int M, int N) {
  __shared__ unsigned short As[16 * 128];

  const int lane  = threadIdx.x & 31;
  const int wave  = threadIdx.x >> 5;
  const int m0    = blockIdx.x * 16;
  const int n0    = blockIdx.y * 128 + wave * 16;
  const int bcol  = n0 + (lane & 15);
  const int khalf = (lane >> 4) << 3;          // 0 | 8

  // hint the B strip into cache (global_prefetch_b8)
  __builtin_prefetch(Wt + (size_t)bcol * 128, 0, 3);

  // stage the A tile via async DMA: thread i moves one 16B chunk to LDS
  {
    const int rowi = threadIdx.x >> 4;         // 0..15
    const int ch   = threadIdx.x & 15;         // 0..15
    const unsigned short* gp = Ah + (size_t)(m0 + rowi) * 128 + ch * 8;
    unsigned lds_off =
        (unsigned)(unsigned long long)(lds_us_t*)(As + rowi * 128 + ch * 8);
    asm volatile("global_load_async_to_lds_b128 %0, %1, off"
                 :: "v"(lds_off), "v"(gp) : "memory");
    asm volatile("s_wait_asynccnt 0x0" ::: "memory");
  }
  __syncthreads();

  const unsigned short* arow = As + (lane & 15) * 128;

  v8f c = {};
  #pragma unroll
  for (int kt = 0; kt < 128; kt += 32) {
    const int kbase = kt + khalf;
    Frag16 a, b;
    a.q[0] = *(const uint4*)(arow + kbase);        // K = kbase .. kbase+7
    a.q[1] = *(const uint4*)(arow + kbase + 16);   // K = kbase+16 .. +23
    const unsigned short* bp = Wt + (size_t)bcol * 128 + kbase;
    b.q[0] = *(const uint4*)(bp);
    b.q[1] = *(const uint4*)(bp + 16);
    c = __builtin_amdgcn_wmma_f32_16x16x32_f16(false, a.v, false, b.v,
                                               (short)0, c, false, false);
  }

  const int orow = m0 + ((lane < 16) ? 0 : 8);
  const float bv = bias[bcol];
  #pragma unroll
  for (int r = 0; r < 8; ++r) {
    float val = c[r] + bv;
    if (ACT == 1) val = fmaxf(val, 0.0f);
    float* p = Out + (size_t)(orow + r) * N + bcol;
    if (ADDTO) val += *p;
    *p = val;
    if (HOUT) Outh[(size_t)(orow + r) * N + bcol] = f16bits(val);
  }
}

// fp32 [K][N] -> f16 bits, transposed [N][K]
__global__ void k_wconv(const float* __restrict__ W, unsigned short* __restrict__ Wt,
                        int K, int N) {
  int i = blockIdx.x * 256 + threadIdx.x;
  if (i >= K * N) return;
  int k = i / N, n = i % N;
  Wt[(size_t)n * K + k] = f16bits(W[i]);
}

// small-K input projection with ReLU (fc_x K=4, fc_t K=8); writes f16 mirror
__global__ void k_dense_small(const float* __restrict__ X, const float* __restrict__ W,
                              const float* __restrict__ b, float* __restrict__ Out,
                              unsigned short* __restrict__ Outh,
                              int M, int K, int N) {
  int i = blockIdx.x * 256 + threadIdx.x;
  if (i >= M * N) return;
  int m = i / N, n = i % N;
  float s = b[n];
  for (int k = 0; k < K; ++k) s += X[(size_t)m * K + k] * W[(size_t)k * N + n];
  s = fmaxf(s, 0.0f);
  Out[i] = s;
  Outh[i] = f16bits(s);
}

__global__ void k_fill_f32(float* p, float v, int n) {
  int i = blockIdx.x * 256 + threadIdx.x;
  if (i < n) p[i] = v;
}
__global__ void k_fill_u32(unsigned* p, unsigned v, int n) {
  int i = blockIdx.x * 256 + threadIdx.x;
  if (i < n) p[i] = v;
}

// pass 1: per-(edge,head) wave computes q.k dot, stores logit, atomicMax seg max
__global__ __launch_bounds__(256)
void k_edge_logits(const float* __restrict__ q, const float* __restrict__ k,
                   const int* __restrict__ src, const int* __restrict__ dst,
                   float* __restrict__ logits, unsigned* __restrict__ mseg, int E) {
  int gw = blockIdx.x * 8 + (threadIdx.x >> 5);
  if (gw >= E * NH) return;
  int lane = threadIdx.x & 31;
  int e = gw >> 2, h = gw & 3;
  int s = src[e], d = dst[e];
  const float* qp = q + (size_t)d * HC + h * CC + lane * 4;
  const float* kp = k + (size_t)s * HC + h * CC + lane * 4;
  float acc = 0.0f;
  #pragma unroll
  for (int j = 0; j < 4; ++j) acc += qp[j] * kp[j];
  for (int off = 16; off; off >>= 1) acc += __shfl_xor(acc, off, 32);
  acc *= 0.08838834764831845f;   // 1/sqrt(128)
  if (lane == 0) {
    logits[gw] = acc;
    atomicMax(&mseg[(size_t)d * NH + h], fenc(acc));
  }
}

// decode seg max; empty segments -> 0 (matches reference isfinite fixup)
__global__ void k_seg_m(const unsigned* __restrict__ mseg, float* __restrict__ mf, int n) {
  int i = blockIdx.x * 256 + threadIdx.x;
  if (i >= n) return;
  unsigned u = mseg[i];
  mf[i] = (u == ENC_NEG_INF) ? 0.0f : fdec(u);
}

// pass 2: a=exp(l-m); den += a; msg += a*v[src]
__global__ __launch_bounds__(256)
void k_edge_accum(const float* __restrict__ v, const int* __restrict__ src,
                  const int* __restrict__ dst, const float* __restrict__ logits,
                  const float* __restrict__ mf, float* __restrict__ den,
                  float* __restrict__ msg, int E) {
  int gw = blockIdx.x * 8 + (threadIdx.x >> 5);
  if (gw >= E * NH) return;
  int lane = threadIdx.x & 31;
  int e = gw >> 2, h = gw & 3;
  int s = src[e], d = dst[e];
  float a = __expf(logits[gw] - mf[(size_t)d * NH + h]);
  if (lane == 0) atomicAdd(&den[(size_t)d * NH + h], a);
  const float* vp = v + (size_t)s * HC + h * CC + lane * 4;
  float* mp = msg + (size_t)d * HC + h * CC + lane * 4;
  #pragma unroll
  for (int j = 0; j < 4; ++j) atomicAdd(mp + j, a * vp[j]);
}

// out[i,d] += mean_h msg[i,h,d] / (den[i,h]+eps)   (out already holds skip)
__global__ void k_conv_fin(const float* __restrict__ msg, const float* __restrict__ den,
                           float* __restrict__ out, int Nd) {
  int i = blockIdx.x * 256 + threadIdx.x;
  if (i >= Nd * DD) return;
  int node = i / DD, ch = i % DD;
  float s = 0.0f;
  #pragma unroll
  for (int h = 0; h < NH; ++h)
    s += msg[(size_t)node * HC + h * CC + ch] / (den[(size_t)node * NH + h] + 1e-16f);
  out[i] += 0.25f * s;
}

// tn = relu(tf + ott + oxct), plus f16 mirror
__global__ void k_relu3(const float* a, const float* b, const float* c,
                        float* o, unsigned short* oh, int n) {
  int i = blockIdx.x * 256 + threadIdx.x;
  if (i >= n) return;
  float v = fmaxf(a[i] + b[i] + c[i], 0.0f);
  o[i] = v;
  oh[i] = f16bits(v);
}

// tf = tn + ob, plus f16 mirror
__global__ void k_add2(const float* a, const float* b, float* o,
                       unsigned short* oh, int n) {
  int i = blockIdx.x * 256 + threadIdx.x;
  if (i >= n) return;
  float v = a[i] + b[i];
  o[i] = v;
  oh[i] = f16bits(v);
}

// layernorm over D=128, one wave per row; input already ReLU'd by GEMM epilogue
__global__ __launch_bounds__(256)
void k_layernorm(const float* __restrict__ X, const float* __restrict__ g,
                 const float* __restrict__ beta, float* __restrict__ O, int M) {
  int row = blockIdx.x * 8 + (threadIdx.x >> 5);
  if (row >= M) return;
  int lane = threadIdx.x & 31;
  const float* xr = X + (size_t)row * DD + lane * 4;
  float h0 = xr[0], h1 = xr[1], h2 = xr[2], h3 = xr[3];
  float s = h0 + h1 + h2 + h3;
  for (int off = 16; off; off >>= 1) s += __shfl_xor(s, off, 32);
  float mu = s * (1.0f / 128.0f);
  float d0 = h0 - mu, d1 = h1 - mu, d2 = h2 - mu, d3 = h3 - mu;
  float vs = d0 * d0 + d1 * d1 + d2 * d2 + d3 * d3;
  for (int off = 16; off; off >>= 1) vs += __shfl_xor(vs, off, 32);
  float rs = rsqrtf(vs * (1.0f / 128.0f) + 1e-5f);
  float* orow = O + (size_t)row * DD + lane * 4;
  const float* gp = g + lane * 4;
  const float* bp = beta + lane * 4;
  orow[0] = gp[0] * d0 * rs + bp[0];
  orow[1] = gp[1] * d1 * rs + bp[1];
  orow[2] = gp[2] * d2 * rs + bp[2];
  orow[3] = gp[3] * d3 * rs + bp[3];
}

// final scores: tf @ fc_out.W + b, masked
__global__ void k_scores(const float* __restrict__ tf, const float* __restrict__ W,
                         const float* __restrict__ b, const unsigned char* __restrict__ mask,
                         float* __restrict__ out, int n) {
  int i = blockIdx.x * 256 + threadIdx.x;
  if (i >= n) return;
  float s = b[0];
  for (int k = 0; k < DD; ++k) s += tf[(size_t)i * DD + k] * W[k];
  out[i] = mask[i] ? s : -__builtin_inff();
}

// ---------------------------------------------------------------------------
extern "C" void kernel_launch(void* const* d_in, const int* in_sizes, int n_in,
                              void* d_out, int out_size, void* d_ws, size_t ws_size,
                              hipStream_t stream) {
  const float* x  = (const float*)d_in[0];
  const float* t  = (const float*)d_in[1];
  const int* ets  = (const int*)d_in[2];
  const int* etd  = (const int*)d_in[3];
  const int* exs  = (const int*)d_in[4];
  const int* exd  = (const int*)d_in[5];
  const unsigned char* mask = (const unsigned char*)d_in[6];
  // params pytree leaves, JAX sorted-key dict flattening:
  //   blocks[l]: fcx{W1,W2,b1,b2} out{W,b,beta,g} tt{Wk,Wq,Ws,Wv,bk,bq,bs,bv} xct{...}
  //   then fc_out{W,b} fc_t{W,b} fc_x{W,b}
  auto P = [&](int i) -> const float* { return (const float*)d_in[7 + i]; };
  const int BL = 24;
  const float* fco_W = P(3 * BL + 0);
  const float* fco_b = P(3 * BL + 1);
  const float* fct_W = P(3 * BL + 2);
  const float* fct_b = P(3 * BL + 3);
  const float* fcxW  = P(3 * BL + 4);
  const float* fcxb  = P(3 * BL + 5);

  // ---- workspace bump allocator (256B aligned) ----
  char* wsp = (char*)d_ws;
  size_t off = 0;
  auto alloc = [&](size_t bytes) -> void* {
    off = (off + 255) & ~(size_t)255;
    void* p = wsp + off;
    off += bytes;
    return p;
  };
  float*    xf     = (float*)alloc((size_t)NX * DD * 4);
  float*    tf     = (float*)alloc((size_t)NT * DD * 4);
  float*    tn     = (float*)alloc((size_t)NT * DD * 4);
  float*    ott    = (float*)alloc((size_t)NT * DD * 4);
  float*    oxct   = (float*)alloc((size_t)NT * DD * 4);
  float*    ob     = (float*)alloc((size_t)NT * DD * 4);
  float*    qb     = (float*)alloc((size_t)NT * HC * 4);
  float*    kb     = (float*)alloc((size_t)NX * HC * 4);
  float*    vb     = (float*)alloc((size_t)NX * HC * 4);
  float*    msg    = (float*)alloc((size_t)NT * HC * 4);
  float*    den    = (float*)alloc((size_t)NT * NH * 4);
  float*    mf     = (float*)alloc((size_t)NT * NH * 4);
  unsigned* mseg   = (unsigned*)alloc((size_t)NT * NH * 4);
  float*    logit  = (float*)alloc((size_t)NE * NH * 4);
  float*    hid    = (float*)alloc((size_t)NX * DD * 4);
  // f16 mirrors of GEMM A-side activations
  unsigned short* xfh  = (unsigned short*)alloc((size_t)NX * DD * 2);
  unsigned short* tfh  = (unsigned short*)alloc((size_t)NT * DD * 2);
  unsigned short* tnh  = (unsigned short*)alloc((size_t)NT * DD * 2);
  unsigned short* hidh = (unsigned short*)alloc((size_t)NX * DD * 2);
  unsigned short* wpool = (unsigned short*)alloc((size_t)1500000 * 2);
  unsigned short* wcur = wpool;

  auto wconv = [&](const float* W, int K, int N) -> const unsigned short* {
    unsigned short* dst = wcur;
    wcur += (size_t)K * N;
    int tot = K * N;
    k_wconv<<<(tot + 255) / 256, 256, 0, stream>>>(W, dst, K, N);
    return dst;
  };
  auto gemm = [&](const unsigned short* Ah, const unsigned short* Wt,
                  const float* bias, float* Out, unsigned short* Outh,
                  int M, int N, int relu, int add) {
    dim3 grid(M / 16, N / 128);
    if (Outh == nullptr) {
      if (relu)
        k_gemm<1, 0, 0><<<grid, 256, 0, stream>>>(Ah, Wt, bias, Out, nullptr, M, N);
      else
        k_gemm<0, 0, 0><<<grid, 256, 0, stream>>>(Ah, Wt, bias, Out, nullptr, M, N);
    } else {
      if (add)
        k_gemm<0, 1, 1><<<grid, 256, 0, stream>>>(Ah, Wt, bias, Out, Outh, M, N);
      else
        k_gemm<1, 0, 1><<<grid, 256, 0, stream>>>(Ah, Wt, bias, Out, Outh, M, N);
    }
  };
  auto fill_f = [&](float* p, float v, int n) {
    k_fill_f32<<<(n + 255) / 256, 256, 0, stream>>>(p, v, n);
  };
  auto fill_u = [&](unsigned* p, unsigned v, int n) {
    k_fill_u32<<<(n + 255) / 256, 256, 0, stream>>>(p, v, n);
  };

  // transformer-conv: dst features are always tf/tfh; writes skip+attn to outp
  auto tconv = [&](const unsigned short* srch, int Ns,
                   const int* esrc, const int* edst,
                   const unsigned short* Wq, const float* bq,
                   const unsigned short* Wk, const float* bk,
                   const unsigned short* Wv, const float* bv,
                   const unsigned short* Ws, const float* bs, float* outp) {
    gemm(tfh,  Wq, bq, qb,   nullptr, NT, HC, 0, 0);
    gemm(srch, Wk, bk, kb,   nullptr, Ns, HC, 0, 0);
    gemm(srch, Wv, bv, vb,   nullptr, Ns, HC, 0, 0);
    gemm(tfh,  Ws, bs, outp, nullptr, NT, DD, 0, 0);
    fill_u(mseg, ENC_NEG_INF, NT * NH);
    fill_f(den, 0.0f, NT * NH);
    fill_f(msg, 0.0f, NT * HC);
    int nwaves = NE * NH;
    k_edge_logits<<<nwaves / 8, 256, 0, stream>>>(qb, kb, esrc, edst, logit, mseg, NE);
    k_seg_m<<<(NT * NH + 255) / 256, 256, 0, stream>>>(mseg, mf, NT * NH);
    k_edge_accum<<<nwaves / 8, 256, 0, stream>>>(vb, esrc, edst, logit, mf, den, msg, NE);
    k_conv_fin<<<(NT * DD + 255) / 256, 256, 0, stream>>>(msg, den, outp, NT);
  };

  // ---- input projections (fp32 + f16 mirror) ----
  k_dense_small<<<(NX * DD + 255) / 256, 256, 0, stream>>>(x, fcxW, fcxb, xf, xfh, NX, 4, DD);
  k_dense_small<<<(NT * DD + 255) / 256, 256, 0, stream>>>(t, fct_W, fct_b, tf, tfh, NT, 8, DD);

  // ---- layers ----
  for (int l = 0; l < 3; ++l) {
    int pb = l * BL;
    const float* W1   = P(pb + 0);  const float* W2   = P(pb + 1);
    const float* b1   = P(pb + 2);  const float* b2   = P(pb + 3);
    const float* oW   = P(pb + 4);  const float* obi  = P(pb + 5);
    const float* obt  = P(pb + 6);  const float* og   = P(pb + 7);
    const float* ttWk = P(pb + 8);  const float* ttWq = P(pb + 9);
    const float* ttWs = P(pb + 10); const float* ttWv = P(pb + 11);
    const float* ttbk = P(pb + 12); const float* ttbq = P(pb + 13);
    const float* ttbs = P(pb + 14); const float* ttbv = P(pb + 15);
    const float* xcWk = P(pb + 16); const float* xcWq = P(pb + 17);
    const float* xcWs = P(pb + 18); const float* xcWv = P(pb + 19);
    const float* xcbk = P(pb + 20); const float* xcbq = P(pb + 21);
    const float* xcbs = P(pb + 22); const float* xcbv = P(pb + 23);

    const unsigned short* tWq = wconv(ttWq, DD, HC);
    const unsigned short* tWk = wconv(ttWk, DD, HC);
    const unsigned short* tWv = wconv(ttWv, DD, HC);
    const unsigned short* tWs = wconv(ttWs, DD, DD);
    const unsigned short* xWq = wconv(xcWq, DD, HC);
    const unsigned short* xWk = wconv(xcWk, DD, HC);
    const unsigned short* xWv = wconv(xcWv, DD, HC);
    const unsigned short* xWs = wconv(xcWs, DD, DD);
    const unsigned short* oWt = wconv(oW,   DD, DD);
    const unsigned short* W1t = wconv(W1,   DD, DD);
    const unsigned short* W2t = wconv(W2,   DD, DD);

    tconv(tfh, NT, ets, etd, tWq, ttbq, tWk, ttbk, tWv, ttbv, tWs, ttbs, ott);
    tconv(xfh, NX, exs, exd, xWq, xcbq, xWk, xcbk, xWv, xcbv, xWs, xcbs, oxct);

    k_relu3<<<(NT * DD + 255) / 256, 256, 0, stream>>>(tf, ott, oxct, tn, tnh, NT * DD);

    gemm(tnh, oWt, obi, qb, nullptr, NT, DD, 1, 0);               // relu(tn@W+b) -> qb
    k_layernorm<<<NT / 8, 256, 0, stream>>>(qb, og, obt, ob, NT); // ob = LN(...)

    gemm(xfh, W1t, b1, hid, hidh, NX, DD, 1, 0);                  // hid = relu(xf@W1+b1)
    gemm(hidh, W2t, b2, xf, xfh, NX, DD, 0, 1);                   // xf += hid@W2+b2

    k_add2<<<(NT * DD + 255) / 256, 256, 0, stream>>>(tn, ob, tf, tfh, NT * DD);
  }

  // ---- final scores ----
  k_scores<<<(NT + 255) / 256, 256, 0, stream>>>(tf, fco_W, fco_b, mask,
                                                 (float*)d_out, NT);
  (void)in_sizes; (void)n_in; (void)out_size; (void)ws_size;
}
